// Decoder_85916525789418
// MI455X (gfx1250) — matrix operand order repaired and checked by reading, not compiled, not run on hardware
//
#include <hip/hip_runtime.h>

typedef _Float16 v8h  __attribute__((ext_vector_type(8)));
typedef _Float16 v16h __attribute__((ext_vector_type(16)));
typedef float    v8f  __attribute__((ext_vector_type(8)));
typedef float    v2f  __attribute__((ext_vector_type(2)));

#define B_TOT   65536
#define T_STEPS 12
#define G3      384   // 3*H
#define NWAVES  8

#define WMMA_F16(a,b,c) __builtin_amdgcn_wmma_f32_16x16x32_f16(false,(a),false,(b),(short)0,(c),false,false)
#define WMMA_F32(a,b,c) __builtin_amdgcn_wmma_f32_16x16x4_f32 (false,(a),false,(b),(short)0,(c),false,false)

// ---- fragment loaders (layouts per CDNA5 ISA §7.12.2) ----
// A (16x32 f16): lane m=lane&15; lanes0-15 hold K={k0..k0+7, k0+16..k0+23}, lanes16-31 shifted by 8.
__device__ __forceinline__ v16h ldfragA(const _Float16* p, int ldk, int k0, int lane) {
  const _Float16* q = p + (lane & 15) * ldk + k0 + ((lane & 16) ? 8 : 0);
  v8h lo = *(const v8h*)q;
  v8h hi = *(const v8h*)(q + 16);
  return __builtin_shufflevector(lo, hi, 0,1,2,3,4,5,6,7,8,9,10,11,12,13,14,15);
}
// B (32x16 f16) from [N][K] row-major: lane n=lane&15; lanes0-15 hold K={k0..k0+7, k0+16..k0+23}, lanes16-31 shifted by 8.
__device__ __forceinline__ v16h ldfragB(const _Float16* p, int ldk, int nt, int k0, int lane) {
  const _Float16* q = p + (size_t)(nt * 16 + (lane & 15)) * ldk + k0 + ((lane & 16) ? 8 : 0);
  v8h lo = *(const v8h*)q;
  v8h hi = *(const v8h*)(q + 16);
  return __builtin_shufflevector(lo, hi, 0,1,2,3,4,5,6,7,8,9,10,11,12,13,14,15);
}

__device__ __forceinline__ float sigm(float x) { return 1.f / (1.f + __expf(-x)); }

// ---------------- weight prep: fp32 -> f16, packed layouts ----------------
__global__ __launch_bounds__(256) void prep_kernel(
    const float* __restrict__ Wdec, const float* __restrict__ Wih,
    const float* __restrict__ Whh,  const float* __restrict__ Wattn,
    const float* __restrict__ Wmu,  const float* __restrict__ Wstd,
    _Float16* __restrict__ out) {
  int i = blockIdx.x * 256 + threadIdx.x;
  const int n0 = 128 * 288, n1 = 384 * 288, n2 = 384 * 128, n3 = 128 * 256, n4 = 16 * 128;
  if (i < n0) { out[i] = (_Float16)Wdec[i]; return; }
  i -= n0; _Float16* o = out + n0;
  if (i < n1) { int r = i / 288, c = i - r * 288; o[i] = (_Float16)Wih[r * 290 + c]; return; }
  i -= n1; o += n1;
  if (i < n2) { o[i] = (_Float16)Whh[i]; return; }
  i -= n2; o += n2;
  if (i < n3) { o[i] = (_Float16)Wattn[i]; return; }
  i -= n3; o += n3;
  if (i < n4) {
    int r = i / 128, c = i - r * 128;
    float v = (r < 2) ? Wmu[r * 128 + c] : ((r < 4) ? Wstd[(r - 2) * 128 + c] : 0.f);
    o[i] = (_Float16)v;
  }
}

// ---------------- main decoder: 1 block = 1 sequence of 16 rows ----------------
__global__ __launch_bounds__(256) void decoder_kernel(
    const float* __restrict__ last_state, const float* __restrict__ enc,
    const float* __restrict__ zin,        const float* __restrict__ fut,
    const float* __restrict__ b_dec,  const float* __restrict__ W_vel,
    const float* __restrict__ b_vel,  const float* __restrict__ W_ih_full,
    const float* __restrict__ b_ih,   const float* __restrict__ b_hh,
    const float* __restrict__ b_attn, const float* __restrict__ b_mu,
    const float* __restrict__ b_std,
    const _Float16* __restrict__ Wdec16, const _Float16* __restrict__ Wih16,
    const _Float16* __restrict__ Whh16,  const _Float16* __restrict__ Wattn16,
    const _Float16* __restrict__ Wms16,  float* __restrict__ out) {
  __shared__ _Float16 zx16[16 * 296];   // zx (f16), ld=296 (+8 pad)
  __shared__ _Float16 h16[16 * 136];    // h (f16), ld=136
  __shared__ _Float16 ctx16[16 * 136];  // ctx (f16)
  __shared__ float    h32[16 * 132];    // h (f32), ld=132
  __shared__ float    gx32[16 * 388];   // gx_zx + b_ih (+b_hh r/z), ld=388
  __shared__ float    wiaS[G3 * 2];     // W_ih[:,288:290]
  __shared__ float    aS[16 * 2];       // current a (carry)
  __shared__ float    attnS[16 * 20];   // softmax probs, ld=20

  const int tid  = threadIdx.x;
  const int lane = tid & 31;
  const int wave = tid >> 5;                // 0..7
  const int s16  = blockIdx.x * 16;         // first global row of this sequence
  const int nloc = lane & 15;
  const int ncol = wave * 16 + nloc;        // hidden column owned by this lane
  const int mrow = (lane >> 4) * 8;         // D-fragment row base

  // ---- stage zx, W_ia, a0 ----
  for (int e = tid; e < 16 * 288; e += 256) {
    int m = e / 288, c = e - m * 288;
    float v = (c < 256) ? enc[(s16 + m) * 256 + c] : zin[(s16 + m) * 32 + (c - 256)];
    zx16[m * 296 + c] = (_Float16)v;
  }
  for (int e = tid; e < G3 * 2; e += 256) {
    int c = e >> 1, j = e & 1;
    wiaS[e] = W_ih_full[c * 290 + 288 + j];
  }
  if (tid < 32) {
    int m = tid >> 1, n = tid & 1;
    float s = b_vel[n];
    for (int k = 0; k < 6; ++k) s += last_state[(s16 + m) * 6 + k] * W_vel[n * 6 + k];
    aS[m * 2 + n] = s;
  }
  __syncthreads();

  // ---- h0 = zx @ Wdec^T + b_dec ----
  {
    v8f acc = {};
    for (int kt = 0; kt < 9; ++kt) {
      v16h a = ldfragA(zx16, 296, kt * 32, lane);
      v16h b = ldfragB(Wdec16, 288, wave, kt * 32, lane);
      acc = WMMA_F16(a, b, acc);
    }
    float bd = b_dec[ncol];
    for (int j = 0; j < 8; ++j) {
      int m = mrow + j;
      float v = acc[j] + bd;
      h32[m * 132 + ncol] = v;
      h16[m * 136 + ncol] = (_Float16)v;
    }
  }
  // ---- gx_zx = zx @ Wih[:, :288]^T + b_ih (+ b_hh folded for r,z parts) ----
  for (int g = 0; g < 3; ++g) {
    int nt = wave + 8 * g;
    v8f acc = {};
    for (int kt = 0; kt < 9; ++kt) {
      v16h a = ldfragA(zx16, 296, kt * 32, lane);
      v16h b = ldfragB(Wih16, 288, nt, kt * 32, lane);
      acc = WMMA_F16(a, b, acc);
    }
    int col = nt * 16 + nloc;
    float bias = b_ih[col] + ((col < 256) ? b_hh[col] : 0.f);
    for (int j = 0; j < 8; ++j) gx32[(mrow + j) * 388 + col] = acc[j] + bias;
  }
  __syncthreads();

  const float bhn = b_hh[256 + ncol];
  const float bat = b_attn[ncol];
  float bms = 0.f;
  if (nloc < 2) bms = b_mu[nloc]; else if (nloc < 4) bms = b_std[nloc - 2];

  for (int t = 0; t < T_STEPS; ++t) {
    // (1) gh = h @ Whh^T   (3 gate tiles per wave, K=128)
    v8f aR = {}, aZ = {}, aN = {};
    for (int kt = 0; kt < 4; ++kt) {
      v16h af = ldfragA(h16, 136, kt * 32, lane);
      v16h bR = ldfragB(Whh16, 128, wave,      kt * 32, lane);
      v16h bZ = ldfragB(Whh16, 128, 8 + wave,  kt * 32, lane);
      v16h bN = ldfragB(Whh16, 128, 16 + wave, kt * 32, lane);
      aR = WMMA_F16(af, bR, aR);
      aZ = WMMA_F16(af, bZ, aZ);
      aN = WMMA_F16(af, bN, aN);
    }
    __syncthreads();
    // GRU gates (fp32 VALU), write h
    for (int j = 0; j < 8; ++j) {
      int m = mrow + j;
      float a0 = aS[m * 2], a1 = aS[m * 2 + 1];
      int c = ncol;
      float gr = gx32[m * 388 + c]       + a0 * wiaS[c * 2]           + a1 * wiaS[c * 2 + 1];
      float gz = gx32[m * 388 + 128 + c] + a0 * wiaS[(128 + c) * 2]   + a1 * wiaS[(128 + c) * 2 + 1];
      float gn = gx32[m * 388 + 256 + c] + a0 * wiaS[(256 + c) * 2]   + a1 * wiaS[(256 + c) * 2 + 1];
      float r   = sigm(gr + aR[j]);
      float zg  = sigm(gz + aZ[j]);
      float pre = gn + r * (aN[j] + bhn);
      float ng  = 1.f - 2.f / (__expf(2.f * pre) + 1.f);   // tanh
      float hold = h32[m * 132 + c];
      float hn = (1.f - zg) * ng + zg * hold;
      h32[m * 132 + c] = hn;
      h16[m * 136 + c] = (_Float16)hn;
    }
    __syncthreads();
    // (2) scores = h h^T (Gram: A==B fragments) + row softmax  -- wave 0
    if (wave == 0) {
      v8f sc = {};
      for (int kt = 0; kt < 4; ++kt) {
        v16h af = ldfragA(h16, 136, kt * 32, lane);
        v16h bf = ldfragB(h16, 136, 0, kt * 32, lane);
        sc = WMMA_F16(af, bf, sc);
      }
      for (int j = 0; j < 8; ++j) {
        float v = sc[j];
        float mx = v;
        for (int msk = 1; msk <= 8; msk <<= 1) mx = fmaxf(mx, __shfl_xor(mx, msk, 32));
        float e = __expf(v - mx);
        float sm = e;
        for (int msk = 1; msk <= 8; msk <<= 1) sm += __shfl_xor(sm, msk, 32);
        attnS[(mrow + j) * 20 + nloc] = e / sm;
      }
    }
    __syncthreads();
    // (3) ctx = attn @ h  (fp32 WMMA 16x16x4, K=16)
    {
      v8f cacc = {};
      int kb = (lane & 16) ? 2 : 0;
      for (int kt = 0; kt < 4; ++kt) {
        int k0 = kt * 4 + kb;
        v2f av, bv;
        av.x = attnS[(lane & 15) * 20 + k0];
        av.y = attnS[(lane & 15) * 20 + k0 + 1];
        bv.x = h32[k0 * 132 + ncol];
        bv.y = h32[(k0 + 1) * 132 + ncol];
        cacc = WMMA_F32(av, bv, cacc);
      }
      for (int j = 0; j < 8; ++j) ctx16[(mrow + j) * 136 + ncol] = (_Float16)cacc[j];
    }
    __syncthreads();
    // (4) h = [h, ctx] @ Wattn^T + b_attn  (K=256)
    v8f hp = {};
    for (int kt = 0; kt < 4; ++kt) {
      v16h af = ldfragA(h16, 136, kt * 32, lane);
      v16h bf = ldfragB(Wattn16, 256, wave, kt * 32, lane);
      hp = WMMA_F16(af, bf, hp);
    }
    for (int kt = 0; kt < 4; ++kt) {
      v16h af = ldfragA(ctx16, 136, kt * 32, lane);
      v16h bf = ldfragB(Wattn16, 256, wave, 128 + kt * 32, lane);
      hp = WMMA_F16(af, bf, hp);
    }
    __syncthreads();   // all reads of old h16 complete
    for (int j = 0; j < 8; ++j) {
      int m = mrow + j;
      float v = hp[j] + bat;
      h32[m * 132 + ncol] = v;
      h16[m * 136 + ncol] = (_Float16)v;
    }
    if (tid < 32) {  // a <- fut_state[t] for next step
      int m = tid >> 1, jj = tid & 1;
      aS[tid] = fut[((size_t)t * B_TOT + (s16 + m)) * 2 + jj];
    }
    __syncthreads();
    // (5) mu/std via one padded WMMA tile -- wave 0
    if (wave == 0) {
      v8f ms = {};
      for (int kt = 0; kt < 4; ++kt) {
        v16h af = ldfragA(h16, 136, kt * 32, lane);
        v16h bf = ldfragB(Wms16, 128, 0, kt * 32, lane);
        ms = WMMA_F16(af, bf, ms);
      }
      if (nloc < 4) {
        for (int j = 0; j < 8; ++j) {
          int r = s16 + mrow + j;
          float v = ms[j] + bms;
          if (nloc < 2)
            out[(size_t)t * B_TOT * 2 + (size_t)r * 2 + nloc] = v;
          else
            out[(size_t)T_STEPS * B_TOT * 2 + (size_t)t * B_TOT * 2 + (size_t)r * 2 + (nloc - 2)] =
                __expf(0.5f * v);
        }
      }
    }
  }
}

extern "C" void kernel_launch(void* const* d_in, const int* in_sizes, int n_in,
                              void* d_out, int out_size, void* d_ws, size_t ws_size,
                              hipStream_t stream) {
  (void)in_sizes; (void)n_in; (void)out_size; (void)ws_size;
  const float* last_state = (const float*)d_in[0];
  const float* enc        = (const float*)d_in[1];
  const float* zin        = (const float*)d_in[2];
  // d_in[3] seq_start_end (uniform arange partition) and d_in[4] fut_obst are unused by the reference math
  const float* fut        = (const float*)d_in[5];
  const float* W_dec  = (const float*)d_in[6];  const float* b_dec  = (const float*)d_in[7];
  const float* W_vel  = (const float*)d_in[8];  const float* b_vel  = (const float*)d_in[9];
  const float* W_ih   = (const float*)d_in[10]; const float* b_ih   = (const float*)d_in[11];
  const float* W_hh   = (const float*)d_in[12]; const float* b_hh   = (const float*)d_in[13];
  const float* W_attn = (const float*)d_in[14]; const float* b_attn = (const float*)d_in[15];
  const float* W_mu   = (const float*)d_in[16]; const float* b_mu   = (const float*)d_in[17];
  const float* W_std  = (const float*)d_in[18]; const float* b_std  = (const float*)d_in[19];

  _Float16* ws      = (_Float16*)d_ws;
  _Float16* Wdec16  = ws;                       // 128*288
  _Float16* Wih16   = Wdec16  + 128 * 288;      // 384*288
  _Float16* Whh16   = Wih16   + 384 * 288;      // 384*128
  _Float16* Wattn16 = Whh16   + 384 * 128;      // 128*256
  _Float16* Wms16   = Wattn16 + 128 * 256;      // 16*128 (rows 0-1 Wmu, 2-3 Wstd, rest 0)

  const int total_f16 = 128*288 + 384*288 + 384*128 + 128*256 + 16*128;  // 231424
  prep_kernel<<<(total_f16 + 255) / 256, 256, 0, stream>>>(W_dec, W_ih, W_hh, W_attn, W_mu, W_std, ws);

  decoder_kernel<<<B_TOT / 16, 256, 0, stream>>>(
      last_state, enc, zin, fut,
      b_dec, W_vel, b_vel, W_ih, b_ih, b_hh, b_attn, b_mu, b_std,
      Wdec16, Wih16, Whh16, Wattn16, Wms16, (float*)d_out);
}
